// YuanAttention_9483287790028
// MI455X (gfx1250) — matrix operand order repaired and checked by reading, not compiled, not run on hardware
//
#include <hip/hip_runtime.h>
#include <hip/hip_bf16.h>

// ---------------------------------------------------------------------------
// YuanAttention for MI455X (gfx1250).  bf16 WMMA (V_WMMA_F32_16X16X32_BF16),
// fp32 accumulate.  Compute-bound (~450 GFLOP vs ~30us of HBM time), so all
// large GEMMs + attention run on the matrix cores in bf16.  A/K tiles are
// staged to LDS with GLOBAL_LOAD_ASYNC_TO_LDS_B128 (ASYNCcnt), overlapping
// with the manual B/V transpose staging.
// ---------------------------------------------------------------------------

#define Bc      2
#define Lc      1024
#define Dc      4096
#define Hc      32
#define HDc     128
#define Tc      (Bc * Lc)        // 2048
#define Tbc     (Tc + Bc)        // 2050
#define CACHEN  4096
#define D2c     (Dc / 2)         // 2048
#define QKN     (2 * Hc * HDc)   // 8192

typedef __attribute__((ext_vector_type(16))) __bf16 v16bf;
typedef __attribute__((ext_vector_type(8)))  __bf16 v8bf;
typedef __attribute__((ext_vector_type(8)))  float  v8f;

// async-to-LDS builtin takes pointers to <4 x i32> in AS1 (global) / AS3 (LDS)
typedef int v4i_g __attribute__((vector_size(16)));
typedef __attribute__((address_space(1))) v4i_g* gv4p;
typedef __attribute__((address_space(3))) v4i_g* lv4p;

static __device__ __forceinline__ v16bf cat16(v8bf a, v8bf b) {
    return __builtin_shufflevector(a, b, 0,1,2,3,4,5,6,7,8,9,10,11,12,13,14,15);
}
static __device__ __forceinline__ v8f wmma_bf16(v16bf a, v16bf b, v8f c) {
    // D = A(16x32 bf16) * B(32x16 bf16) + C(16x16 f32)
    return __builtin_amdgcn_wmma_f32_16x16x32_bf16(false, a, false, b, (short)0, c, false, false);
}

// 16B global->LDS stage: async (ASYNCcnt) when the toolchain exposes it,
// else synchronous vector copy.  stage_wait() drains this wave's async ops;
// follow with __syncthreads() so all waves' LDS writes are visible.
static __device__ __forceinline__ void stage16(const __bf16* g, __bf16* l) {
#if __has_builtin(__builtin_amdgcn_global_load_async_to_lds_b128)
    __builtin_amdgcn_global_load_async_to_lds_b128(
        (gv4p)(unsigned long long)(size_t)g,
        (lv4p)(unsigned)(size_t)l, 0, 0);
#else
    *(v8bf*)l = *(const v8bf*)g;
#endif
}
static __device__ __forceinline__ void stage_wait() {
#if __has_builtin(__builtin_amdgcn_s_wait_asynccnt)
    __builtin_amdgcn_s_wait_asynccnt(0);
#endif
}

// ---------------------------------------------------------------------------
// fp32 -> bf16 bulk convert
// ---------------------------------------------------------------------------
__global__ __launch_bounds__(256) void k_f32_to_bf16(const float* __restrict__ s,
                                                     __bf16* __restrict__ d, int n) {
    int i = blockIdx.x * 256 + threadIdx.x;
    if (i < n) d[i] = (__bf16)s[i];
}

// ---------------------------------------------------------------------------
// build xt (Tb rows): rows inputs_loc[i] <- x[i]; rows input_lf_loc[b] <- cache[pre[b]]
// ---------------------------------------------------------------------------
__global__ __launch_bounds__(256) void k_build_xt(const float* __restrict__ x,
                                                  const float* __restrict__ cache,
                                                  const int* __restrict__ inputs_loc,
                                                  const int* __restrict__ input_lf_loc,
                                                  const int* __restrict__ pre_lf,
                                                  __bf16* __restrict__ xt, int d, int Trows) {
    int i = blockIdx.x;
    const float* src;
    int dstrow;
    if (i < Trows) { src = x + (size_t)i * d; dstrow = inputs_loc[i]; }
    else { int b = i - Trows; dstrow = input_lf_loc[b]; src = cache + (size_t)pre_lf[b] * d; }
    __bf16* dst = xt + (size_t)dstrow * d;
    for (int j = threadIdx.x; j < d; j += 256) dst[j] = (__bf16)src[j];
}

// ---------------------------------------------------------------------------
// GEMM: C[M,N] = A[M,K](bf16) * B[K,N](bf16), fp32 accum; store f32 or bf16.
// 128x128x32 block tile, 256 threads = 8 waves in 2(M) x 4(N), wave = 64x32.
// A staged async to LDS (row-clamped, branch-free); B staged transposed.
// ---------------------------------------------------------------------------
#define GLDA 40
#define GLDB 40
__global__ __launch_bounds__(256) void k_gemm_bf16(const __bf16* __restrict__ A,
                                                   const __bf16* __restrict__ Bm,
                                                   float* __restrict__ Cf,
                                                   __bf16* __restrict__ Cb,
                                                   int M, int N, int K) {
    __shared__ __attribute__((aligned(16))) __bf16 sA[128 * GLDA];
    __shared__ __attribute__((aligned(16))) __bf16 sB[128 * GLDB]; // transposed: [n][k]
    const int tid = threadIdx.x, wave = tid >> 5, lane = tid & 31;
    const int lm = lane & 15, hf = lane >> 4;
    const int wm = wave >> 2, wn = wave & 3;
    const int m0 = blockIdx.y * 128, n0 = blockIdx.x * 128;

    // branch-free A staging: clamp OOB rows to M-1 (their C rows are never stored)
    const int ch0 = tid * 2;
    int ra0 = m0 + (ch0 >> 2);       if (ra0 >= M) ra0 = M - 1;
    int ra1 = m0 + ((ch0 + 1) >> 2); if (ra1 >= M) ra1 = M - 1;
    const int ka0 = (ch0 & 3) * 8, ka1 = ((ch0 + 1) & 3) * 8;
    __bf16* sa0 = &sA[(ch0 >> 2) * GLDA + ka0];
    __bf16* sa1 = &sA[((ch0 + 1) >> 2) * GLDA + ka1];

    v8f acc[4][2];
#pragma unroll
    for (int mt = 0; mt < 4; ++mt)
#pragma unroll
        for (int nt = 0; nt < 2; ++nt)
#pragma unroll
            for (int i = 0; i < 8; ++i) acc[mt][nt][i] = 0.f;

    for (int k0 = 0; k0 < K; k0 += 32) {
        __syncthreads();
        // kick off async A tile 128x32 (row major)
        stage16(A + (size_t)ra0 * K + k0 + ka0, sa0);
        stage16(A + (size_t)ra1 * K + k0 + ka1, sa1);
        // stage B tile 32x128 transposed -> sB[n][k] (overlaps with async A)
#pragma unroll
        for (int cc = 0; cc < 2; ++cc) {
            int ch = tid * 2 + cc;
            int kk = ch >> 4;                // k row 0..31
            int nc = (ch & 15) * 8;          // n chunk
            v8bf dv = *(const v8bf*)(Bm + (size_t)(k0 + kk) * N + n0 + nc);
            if (k0 + 32 < K) __builtin_prefetch(Bm + (size_t)(k0 + 32 + kk) * N + n0 + nc, 0, 1);
#pragma unroll
            for (int j = 0; j < 8; ++j) sB[(nc + j) * GLDB + kk] = dv[j];
        }
        stage_wait();
        __syncthreads();

        v16bf af[4], bfv[2];
#pragma unroll
        for (int mt = 0; mt < 4; ++mt) {
            // A-operand lane layout: m = lane%16; elems0-7 = K[hf*8..], elems8-15 = K[16+hf*8..]
            const __bf16* ar = &sA[(wm * 64 + mt * 16 + lm) * GLDA];
            v8bf lo = *(const v8bf*)(ar + hf * 8);
            v8bf hi = *(const v8bf*)(ar + 16 + hf * 8);
            af[mt] = cat16(lo, hi);
        }
#pragma unroll
        for (int nt = 0; nt < 2; ++nt) {
            // B-operand lane layout: n = lane%16; elems0-7 = K[hf*8..], elems8-15 = K[16+hf*8..]
            const __bf16* br = &sB[(wn * 32 + nt * 16 + lm) * GLDB];
            v8bf lo = *(const v8bf*)(br + hf * 8);
            v8bf hi = *(const v8bf*)(br + 16 + hf * 8);
            bfv[nt] = cat16(lo, hi);
        }
#pragma unroll
        for (int mt = 0; mt < 4; ++mt)
#pragma unroll
            for (int nt = 0; nt < 2; ++nt)
                acc[mt][nt] = wmma_bf16(af[mt], bfv[nt], acc[mt][nt]);
    }

#pragma unroll
    for (int mt = 0; mt < 4; ++mt)
#pragma unroll
        for (int nt = 0; nt < 2; ++nt) {
            int col = n0 + wn * 32 + nt * 16 + lm;
#pragma unroll
            for (int r = 0; r < 8; ++r) {
                int row = m0 + wm * 64 + mt * 16 + r + hf * 8;
                if (row < M) {
                    if (Cf) Cf[(size_t)row * N + col] = acc[mt][nt][r];
                    else    Cb[(size_t)row * N + col] = (__bf16)acc[mt][nt][r];
                }
            }
        }
}

// ---------------------------------------------------------------------------
// o1[t,j] = c[row, j] + c[row+1, dOut + j], row = outputs_loc[t]  (fp32)
// ---------------------------------------------------------------------------
__global__ __launch_bounds__(256) void k_combine(const float* __restrict__ c,
                                                 const int* __restrict__ outputs_loc,
                                                 float* __restrict__ out, int dOut, int N) {
    int t = blockIdx.x;
    int row = outputs_loc[t];
    for (int j = threadIdx.x; j < dOut; j += 256)
        out[(size_t)t * dOut + j] = c[(size_t)row * N + j] + c[(size_t)(row + 1) * N + dOut + j];
}

// ---------------------------------------------------------------------------
// h = rmsnorm(combine(c2) + hidden) * lnw  -> bf16
// ---------------------------------------------------------------------------
__global__ __launch_bounds__(256) void k_o2_rmsnorm(const float* __restrict__ c2,
                                                    const float* __restrict__ hidden,
                                                    const float* __restrict__ lnw,
                                                    const int* __restrict__ outputs_loc,
                                                    __bf16* __restrict__ hbf) {
    __shared__ float vals[Dc];
    __shared__ float red[8];
    int t = blockIdx.x;
    int row = outputs_loc[t];
    int tid = threadIdx.x, wave = tid >> 5, lane = tid & 31;
    float ss = 0.f;
    for (int j = tid; j < Dc; j += 256) {
        float v = c2[(size_t)row * QKN + j] + c2[(size_t)(row + 1) * QKN + Dc + j]
                + hidden[(size_t)t * Dc + j];
        vals[j] = v;
        ss += v * v;
    }
    for (int off = 16; off >= 1; off >>= 1) ss += __shfl_xor(ss, off, 32);
    if (lane == 0) red[wave] = ss;
    __syncthreads();
    if (tid == 0) {
        float tot = 0.f;
        for (int i = 0; i < 8; ++i) tot += red[i];
        red[0] = tot;
    }
    __syncthreads();
    float inv = rsqrtf(red[0] / (float)Dc + 1e-6f);
    for (int j = tid; j < Dc; j += 256)
        hbf[(size_t)t * Dc + j] = (__bf16)(vals[j] * inv * lnw[j]);
}

// ---------------------------------------------------------------------------
// RoPE on qk (T, H, 2*HD) fp32 -> q_bf/k_bf (T, H*HD) bf16.  q pre-scaled by
// 1/sqrt(HD) so attention scores need no extra scale.
// ---------------------------------------------------------------------------
__global__ __launch_bounds__(256) void k_rope(const float* __restrict__ qk,
                                              const float* __restrict__ rot,
                                              const int* __restrict__ positions,
                                              __bf16* __restrict__ qbf,
                                              __bf16* __restrict__ kbf) {
    int idx = blockIdx.x * 256 + threadIdx.x;
    if (idx >= Tc * Hc * HDc) return;
    int d = idx & (HDc - 1);
    int hh = (idx >> 7) & (Hc - 1);
    int t = idx >> 12;
    int pos = positions[t];
    float f = rot[(size_t)pos * HDc + d];
    float cf = __cosf(f), sf = __sinf(f);
    const float* base = qk + (size_t)t * QKN + hh * (2 * HDc);
    float qv = base[d], kv = base[HDc + d];
    float qr = (d < HDc / 2) ? -base[d + HDc / 2] : base[d - HDc / 2];
    float kr = (d < HDc / 2) ? -base[HDc + d + HDc / 2] : base[HDc + d - HDc / 2];
    float qo = (qv * cf + qr * sf) * 0.08838834764831845f; // 128^-0.5
    float ko = kv * cf + kr * sf;
    size_t o = (size_t)t * (Hc * HDc) + hh * HDc + d;
    qbf[o] = (__bf16)qo;
    kbf[o] = (__bf16)ko;
}

// ---------------------------------------------------------------------------
// Flash attention: block = (qt, h, b); 256 threads = 8 waves x 16 q-rows.
// Online softmax; S and O via WMMA; P repacked C-layout -> A-layout via LDS.
// K tile staged async; V tile transposed manually (overlaps async K).
// ---------------------------------------------------------------------------
#define KLD 136   // K tile LDS row stride (hd), bf16 units
#define VLD 40    // V tile LDS row stride (key), transposed [hd][key]
#define PLD 40
__global__ __launch_bounds__(256) void k_attn(const __bf16* __restrict__ qg,
                                              const __bf16* __restrict__ kg,
                                              const __bf16* __restrict__ vg,
                                              __bf16* __restrict__ og) {
    __shared__ __attribute__((aligned(16))) __bf16 kl[32 * KLD];   // [key][hd]
    __shared__ __attribute__((aligned(16))) __bf16 vl[128 * VLD];  // [hd][key]
    __shared__ __attribute__((aligned(16))) __bf16 pl[8 * 16 * PLD];
    const int qt = blockIdx.x, hh = blockIdx.y, b = blockIdx.z;
    const int tid = threadIdx.x, wave = tid >> 5, lane = tid & 31;
    const int lm = lane & 15, hf = lane >> 4;
    const int qrow0 = qt * 128 + wave * 16;      // q row (within L) for m=0 of this wave

    // Load Q fragments (A operand), 4 chunks of K=32 over HD=128
    v16bf qf[4];
    {
        const __bf16* qrow = qg + (size_t)(b * Lc + qrow0 + lm) * (Hc * HDc) + hh * HDc;
#pragma unroll
        for (int kc = 0; kc < 4; ++kc) {
            v8bf lo = *(const v8bf*)(qrow + kc * 32 + hf * 8);
            v8bf hi = *(const v8bf*)(qrow + kc * 32 + 16 + hf * 8);
            qf[kc] = cat16(lo, hi);
        }
    }

    v8f acc[8];
#pragma unroll
    for (int nt = 0; nt < 8; ++nt)
#pragma unroll
        for (int i = 0; i < 8; ++i) acc[nt][i] = 0.f;
    float mrow[8], lrow[8];
#pragma unroll
    for (int r = 0; r < 8; ++r) { mrow[r] = -3e38f; lrow[r] = 0.f; }

    const int nkt = (qt + 1) * 4;   // causal: key tiles of 32
    for (int kt = 0; kt < nkt; ++kt) {
        __syncthreads();
        // stage K async (key-major) and V sync (transposed, hd-major)
#pragma unroll
        for (int cc = 0; cc < 2; ++cc) {
            int ch = tid * 2 + cc;     // 0..511
            int key = ch >> 4;
            int nc = (ch & 15) * 8;
            size_t gro = (size_t)(b * Lc + kt * 32 + key) * (Hc * HDc) + hh * HDc + nc;
            stage16(kg + gro, &kl[key * KLD + nc]);
            v8bf vd = *(const v8bf*)(vg + gro);
#pragma unroll
            for (int j = 0; j < 8; ++j) vl[(nc + j) * VLD + key] = vd[j];
        }
        stage_wait();
        __syncthreads();

        // S = Q * K^T  (two 16x16 tiles: keys [0,16) and [16,32) of this kt)
        v8f s0, s1;
#pragma unroll
        for (int i = 0; i < 8; ++i) { s0[i] = 0.f; s1[i] = 0.f; }
#pragma unroll
        for (int kc = 0; kc < 4; ++kc) {
            const __bf16* kr0 = &kl[lm * KLD];
            const __bf16* kr1 = &kl[(lm + 16) * KLD];
            v16bf b0 = cat16(*(const v8bf*)(kr0 + kc * 32 + hf * 8),
                             *(const v8bf*)(kr0 + kc * 32 + 16 + hf * 8));
            v16bf b1 = cat16(*(const v8bf*)(kr1 + kc * 32 + hf * 8),
                             *(const v8bf*)(kr1 + kc * 32 + 16 + hf * 8));
            s0 = wmma_bf16(qf[kc], b0, s0);
            s1 = wmma_bf16(qf[kc], b1, s1);
        }

        // causal mask + online softmax (row r lives in one 16-lane half)
        int kcol = kt * 32 + lm;
        float p0[8], p1[8];
#pragma unroll
        for (int r = 0; r < 8; ++r) {
            int qrow = qrow0 + r + hf * 8;
            float a0 = (kcol <= qrow) ? s0[r] : -1e9f;
            float a1 = (kcol + 16 <= qrow) ? s1[r] : -1e9f;
            float mx = fmaxf(a0, a1);
            for (int off = 8; off >= 1; off >>= 1) mx = fmaxf(mx, __shfl_xor(mx, off, 32));
            float mnew = fmaxf(mrow[r], mx);
            float alpha = __expf(mrow[r] - mnew);
            float e0 = __expf(a0 - mnew);
            float e1 = __expf(a1 - mnew);
            float rs = e0 + e1;
            for (int off = 8; off >= 1; off >>= 1) rs += __shfl_xor(rs, off, 32);
            lrow[r] = lrow[r] * alpha + rs;
            mrow[r] = mnew;
            p0[r] = e0; p1[r] = e1;
#pragma unroll
            for (int nt = 0; nt < 8; ++nt) acc[nt][r] *= alpha;
        }

        // repack P: C-layout -> A-layout via per-wave LDS staging
#pragma unroll
        for (int r = 0; r < 8; ++r) {
            int m = r + hf * 8;
            pl[wave * 16 * PLD + m * PLD + lm] = (__bf16)p0[r];
            pl[wave * 16 * PLD + m * PLD + lm + 16] = (__bf16)p1[r];
        }
        __syncthreads();   // order LDS (all waves share trip count)
        v16bf pa;
        {
            const __bf16* pr = &pl[wave * 16 * PLD + lm * PLD];
            pa = cat16(*(const v8bf*)(pr + hf * 8), *(const v8bf*)(pr + 16 + hf * 8));
        }
        // O += P * V
#pragma unroll
        for (int nt = 0; nt < 8; ++nt) {
            const __bf16* vr = &vl[(nt * 16 + lm) * VLD];
            v16bf bv = cat16(*(const v8bf*)(vr + hf * 8), *(const v8bf*)(vr + 16 + hf * 8));
            acc[nt] = wmma_bf16(pa, bv, acc[nt]);
        }
    }

    // normalize and store (T, H*HD) bf16
    float rinv[8];
#pragma unroll
    for (int r = 0; r < 8; ++r) rinv[r] = 1.f / lrow[r];
#pragma unroll
    for (int nt = 0; nt < 8; ++nt) {
#pragma unroll
        for (int r = 0; r < 8; ++r) {
            int t = b * Lc + qrow0 + r + hf * 8;
            og[(size_t)t * (Hc * HDc) + hh * HDc + nt * 16 + lm] = (__bf16)(acc[nt][r] * rinv[r]);
        }
    }
}

// ---------------------------------------------------------------------------
// cache row overwrite: new_cache[out_lf_indexs[b]] = src[(b+1)*L - 1]
// (xt[out_lf_loc[b]] is provably the last token row of sequence b)
// ---------------------------------------------------------------------------
__global__ __launch_bounds__(256) void k_cache_rows(const float* __restrict__ src,
                                                    const int* __restrict__ out_lf,
                                                    float* __restrict__ cache_out, int d) {
    int b = blockIdx.x;
    const float* s = src + (size_t)((b + 1) * Lc - 1) * d;
    float* dst = cache_out + (size_t)out_lf[b] * d;
    for (int j = threadIdx.x; j < d; j += 256) dst[j] = s[j];
}

// ---------------------------------------------------------------------------
extern "C" void kernel_launch(void* const* d_in, const int* in_sizes, int n_in,
                              void* d_out, int out_size, void* d_ws, size_t ws_size,
                              hipStream_t stream) {
    (void)in_sizes; (void)n_in; (void)out_size; (void)ws_size;
    const float* hidden       = (const float*)d_in[0];
    const int*   positions    = (const int*)d_in[1];
    const float* rotary       = (const float*)d_in[2];
    const int*   pre_lf       = (const int*)d_in[3];
    const int*   out_lf       = (const int*)d_in[4];
    const int*   input_lf_loc = (const int*)d_in[5];
    const int*   out_lf_loc   = (const int*)d_in[6];  (void)out_lf_loc;
    const int*   inputs_loc   = (const int*)d_in[7];
    const int*   outputs_loc  = (const int*)d_in[8];
    const float* lf1          = (const float*)d_in[9];
    const float* lf2          = (const float*)d_in[10];
    const float* conv1        = (const float*)d_in[11];
    const float* conv2        = (const float*)d_in[12];
    const float* lnw          = (const float*)d_in[13];
    const float* wqk          = (const float*)d_in[14];
    const float* wv           = (const float*)d_in[15];
    const float* wo           = (const float*)d_in[16];

    float* out     = (float*)d_out;                         // (T, D)
    float* lf1_new = out + (size_t)Tc * Dc;                 // (CACHEN, D)
    float* lf2_new = lf1_new + (size_t)CACHEN * Dc;         // (CACHEN, D/2)

    // ---- workspace carve (~445 MB; all regions 256B aligned) ----
    char* ws = (char*)d_ws;
    size_t off = 0;
    auto carve = [&](size_t bytes) -> char* {
        char* p = ws + off;
        off += (bytes + 255) & ~(size_t)255;
        return p;
    };
    __bf16* hidden_bf = (__bf16*)carve((size_t)Tc * Dc * 2);
    __bf16* wv_bf     = (__bf16*)carve((size_t)Dc * Dc * 2);
    __bf16* conv1_bf  = (__bf16*)carve((size_t)Dc * Dc * 2);
    __bf16* conv2_bf  = (__bf16*)carve((size_t)D2c * QKN * 2);
    __bf16* wqk_bf    = (__bf16*)carve((size_t)Dc * QKN * 2);
    __bf16* wo_bf     = (__bf16*)carve((size_t)Dc * Dc * 2);
    __bf16* xt1_bf    = (__bf16*)carve((size_t)Tbc * Dc * 2);
    float*  c1        = (float*)carve((size_t)Tbc * Dc * 4);
    float*  o1        = (float*)carve((size_t)Tc * D2c * 4);
    __bf16* xt2_bf    = (__bf16*)carve((size_t)Tbc * D2c * 2);
    float*  c2        = (float*)carve((size_t)Tbc * QKN * 4);
    float*  qkbuf     = c2;   // alias: c2 dead after rmsnorm, qk GEMM reuses it
    __bf16* h_bf      = (__bf16*)carve((size_t)Tc * Dc * 2);
    __bf16* q_bf      = (__bf16*)carve((size_t)Tc * Dc * 2);
    __bf16* k_bf      = (__bf16*)carve((size_t)Tc * Dc * 2);
    __bf16* v_bf      = (__bf16*)carve((size_t)Tc * Dc * 2);
    __bf16* o_bf      = (__bf16*)carve((size_t)Tc * Dc * 2);

    auto cvt = [&](const float* s, __bf16* d, size_t n) {
        k_f32_to_bf16<<<(unsigned)((n + 255) / 256), 256, 0, stream>>>(s, d, (int)n);
    };
    auto gemm = [&](const __bf16* A, const __bf16* Bm, float* Cf, __bf16* Cb,
                    int M, int N, int K) {
        dim3 g(N / 128, (M + 127) / 128);
        k_gemm_bf16<<<g, 256, 0, stream>>>(A, Bm, Cf, Cb, M, N, K);
    };

    // cache outputs: bulk copy then overwrite B rows (lf2 rows after o1 exists)
    (void)hipMemcpyAsync(lf1_new, lf1, (size_t)CACHEN * Dc * 4, hipMemcpyDeviceToDevice, stream);
    (void)hipMemcpyAsync(lf2_new, lf2, (size_t)CACHEN * D2c * 4, hipMemcpyDeviceToDevice, stream);
    k_cache_rows<<<Bc, 256, 0, stream>>>(hidden, out_lf, lf1_new, Dc);

    // weight / activation bf16 conversion
    cvt(hidden, hidden_bf, (size_t)Tc * Dc);
    cvt(wv,    wv_bf,    (size_t)Dc * Dc);
    cvt(conv1, conv1_bf, (size_t)Dc * Dc);
    cvt(conv2, conv2_bf, (size_t)D2c * QKN);
    cvt(wqk,   wqk_bf,   (size_t)Dc * QKN);
    cvt(wo,    wo_bf,    (size_t)Dc * Dc);

    // v = hidden @ wv  (bf16 store, (T, H*HD))
    gemm(hidden_bf, wv_bf, nullptr, v_bf, Tc, Dc, Dc);

    // conv branch 1
    k_build_xt<<<Tbc, 256, 0, stream>>>(hidden, lf1, inputs_loc, input_lf_loc, pre_lf,
                                        xt1_bf, Dc, Tc);
    gemm(xt1_bf, conv1_bf, c1, nullptr, Tbc, Dc, Dc);
    k_combine<<<Tc, 256, 0, stream>>>(c1, outputs_loc, o1, D2c, Dc);
    k_cache_rows<<<Bc, 256, 0, stream>>>(o1, out_lf, lf2_new, D2c);

    // conv branch 2
    k_build_xt<<<Tbc, 256, 0, stream>>>(o1, lf2, inputs_loc, input_lf_loc, pre_lf,
                                        xt2_bf, D2c, Tc);
    gemm(xt2_bf, conv2_bf, c2, nullptr, Tbc, QKN, D2c);

    // h = rmsnorm(combine(c2) + hidden) * lnw  -> bf16
    k_o2_rmsnorm<<<Tc, 256, 0, stream>>>(c2, hidden, lnw, outputs_loc, h_bf);

    // qk = h @ wqk (fp32, aliases c2 region), then RoPE -> q_bf/k_bf
    gemm(h_bf, wqk_bf, qkbuf, nullptr, Tc, QKN, Dc);
    k_rope<<<(Tc * Hc * HDc) / 256, 256, 0, stream>>>(qkbuf, rotary, positions, q_bf, k_bf);

    // attention
    {
        dim3 g(Lc / 128, Hc, Bc);
        k_attn<<<g, 256, 0, stream>>>(q_bf, k_bf, v_bf, o_bf);
    }

    // out = o @ wo (fp32 straight into d_out)
    gemm(o_bf, wo_bf, out, nullptr, Tc, Dc, Dc);
}
